// SSKNet_66838281060870
// MI455X (gfx1250) — matrix operand order repaired and checked by reading, not compiled, not run on hardware
//
#include <hip/hip_runtime.h>
#include <hip/hip_bf16.h>
#include <math.h>

typedef _Float16 v16h __attribute__((ext_vector_type(16)));
typedef _Float16 v8h  __attribute__((ext_vector_type(8)));
typedef float    v8f  __attribute__((ext_vector_type(8)));

union V16 { v16h v; v8h h[2]; };

__device__ __forceinline__ float aact(float v, int a) {
    if (a == 1) return v > 0.f ? v : 0.f;
    if (a == 2) return v > 0.f ? v : 0.01f * v;
    return v;
}

// CDNA5 async global->LDS helpers (tracked by ASYNCcnt)
__device__ __forceinline__ void async_g2l_b128(uint32_t ldsAddr, uint64_t gAddr) {
    asm volatile("global_load_async_to_lds_b128 %0, %1, off"
                 :: "v"(ldsAddr), "v"(gAddr) : "memory");
}
__device__ __forceinline__ void async_g2l_b32(uint32_t ldsAddr, uint64_t gAddr) {
    asm volatile("global_load_async_to_lds_b32 %0, %1, off"
                 :: "v"(ldsAddr), "v"(gAddr) : "memory");
}
__device__ __forceinline__ void wait_async0() {
    asm volatile("s_wait_asynccnt 0" ::: "memory");
}

// ---------------------------------------------------------------------------
// Spectral attention: p = GAP(x); p2 = p + 1D convs(k=3,5,7, middle rows);
// att = sigmoid(fc2(relu(fc1(p2))))
// ---------------------------------------------------------------------------
__global__ __launch_bounds__(256) void attention_kernel(
    const float* __restrict__ x,
    const float* __restrict__ c3w, const float* __restrict__ c3b,
    const float* __restrict__ c5w, const float* __restrict__ c5b,
    const float* __restrict__ c7w, const float* __restrict__ c7b,
    const float* __restrict__ fc1w, const float* __restrict__ fc1b,
    const float* __restrict__ fc2w, const float* __restrict__ fc2b,
    float* __restrict__ att)
{
    __shared__ float p[200], p2[200], q[12];
    const int b = blockIdx.x, tid = threadIdx.x;
    if (tid < 200) {
        const float* xp = x + ((size_t)b * 200 + tid) * 121;
        float s = 0.f;
        for (int i = 0; i < 121; ++i) s += xp[i];
        p[tid] = s * (1.f / 121.f);
    }
    __syncthreads();
    if (tid < 200) {
        float s3 = c3b[0];
        for (int t = 0; t < 3; ++t) { int c = tid + t - 1; if (c >= 0 && c < 200) s3 += c3w[3 + t] * p[c]; }
        float s5 = c5b[0];
        for (int t = 0; t < 5; ++t) { int c = tid + t - 2; if (c >= 0 && c < 200) s5 += c5w[10 + t] * p[c]; }
        float s7 = c7b[0];
        for (int t = 0; t < 7; ++t) { int c = tid + t - 3; if (c >= 0 && c < 200) s7 += c7w[21 + t] * p[c]; }
        p2[tid] = p[tid] + s3 + s5 + s7;
    }
    __syncthreads();
    if (tid < 12) {
        float s = fc1b[tid];
        const float* wr = fc1w + tid * 200;
        for (int c = 0; c < 200; ++c) s += wr[c] * p2[c];
        q[tid] = s > 0.f ? s : 0.f;
    }
    __syncthreads();
    if (tid < 200) {
        float s = fc2b[tid];
        const float* wr = fc2w + tid * 12;
        for (int j = 0; j < 12; ++j) s += wr[j] * q[j];
        att[(size_t)b * 200 + tid] = 1.f / (1.f + expf(-s));
    }
}

// ---------------------------------------------------------------------------
// Weight pre-swizzle: OIHW f32 -> per-wave WMMA A-fragments (f16).
// frag id = (tap*KT + kt)*MT + mt ; element (frag, lane, h):
//   M = mt*16 + lane%16 ; K = 8*(lane/16) + (h%8) + 16*(h/8) ; C = kt*32 + K
// ---------------------------------------------------------------------------
__global__ __launch_bounds__(256) void swz_kernel(
    const float* __restrict__ W, _Float16* __restrict__ out,
    int COUT, int CIN, int KSZ, int KT, int MT, int total)
{
    int e = blockIdx.x * blockDim.x + threadIdx.x;
    if (e >= total) return;
    const int h = e & 15;
    const int lane = (e >> 4) & 31;
    const int frag = e >> 9;
    const int mt = frag % MT;
    const int kt = (frag / MT) % KT;
    const int tap = frag / (MT * KT);
    const int m = mt * 16 + (lane & 15);
    const int k = 8 * (lane >> 4) + (h & 7) + 16 * (h >> 3);
    const int c = kt * 32 + k;
    float v = 0.f;
    if (m < COUT && c < CIN) v = W[((size_t)m * CIN + c) * KSZ * KSZ + tap];
    out[e] = (_Float16)v;
}

// ---------------------------------------------------------------------------
// Per-channel batch-norm statistics over (B=512, 121): writes scale,bias.
// ---------------------------------------------------------------------------
__global__ __launch_bounds__(256) void stats_kernel(
    const float* __restrict__ buf, int C, float* __restrict__ st)
{
    const int c = blockIdx.x, tid = threadIdx.x;
    float s = 0.f, s2 = 0.f;
    for (int i = tid; i < 512 * 121; i += 256) {
        int b = i / 121, p = i - b * 121;
        float v = buf[((size_t)b * C + c) * 121 + p];
        s += v; s2 += v * v;
    }
    __shared__ float rs[256], rq[256];
    rs[tid] = s; rq[tid] = s2;
    __syncthreads();
    for (int o = 128; o > 0; o >>= 1) {
        if (tid < o) { rs[tid] += rs[tid + o]; rq[tid] += rq[tid + o]; }
        __syncthreads();
    }
    if (tid == 0) {
        const float n = 512.f * 121.f;
        float m = rs[0] / n;
        float var = rq[0] / n - m * m;
        float sc = rsqrtf(var + 1e-5f);
        st[c] = sc;
        st[C + c] = -m * sc;
    }
}

// ---------------------------------------------------------------------------
// Elementwise combine: out = actF( actA(sa*a+ba) + actB(sb*b+bb) )
// ---------------------------------------------------------------------------
__global__ __launch_bounds__(256) void combine_kernel(
    const float* __restrict__ a, const float* __restrict__ sa,
    const float* __restrict__ b2, const float* __restrict__ sb,
    float* __restrict__ out, int C, int total, int actA, int actB, int actF)
{
    for (int i = blockIdx.x * blockDim.x + threadIdx.x; i < total;
         i += gridDim.x * blockDim.x) {
        int c = (i / 121) % C;
        float va = a[i];
        if (sa) va = va * sa[c] + sa[C + c];
        va = aact(va, actA);
        float vb = b2[i];
        if (sb) vb = vb * sb[c] + sb[C + c];
        vb = aact(vb, actB);
        out[i] = aact(va + vb, actF);
    }
}

// ---------------------------------------------------------------------------
// WMMA implicit-GEMM conv (KSZxKSZ, pad PAD, 11x11 spatial), one batch/WG.
// Input staged to LDS as padded [pos][chan] f16 with bn+act folded in.
// Weights (pre-swizzled A-fragments) are double-buffered into LDS with
// GLOBAL_LOAD_ASYNC_TO_LDS_B128: tap t+1 streams in while tap t is in WMMA.
// ---------------------------------------------------------------------------
template<int CIN, int COUT, int KSZ, int PAD, int CHS>
__global__ __launch_bounds__(256) void conv_wmma(
    const float* __restrict__ in, const float* __restrict__ st, int act,
    const _Float16* __restrict__ wf, float* __restrict__ out)
{
    constexpr int TAPS = KSZ * KSZ;
    constexpr int KT = (CIN + 31) / 32;
    constexpr int MT = (COUT + 15) / 16;
    constexpr int WP = 11 + 2 * PAD;
    constexpr int NPOS = WP * WP;
    constexpr int INH = ((NPOS * CHS) + 63) & ~63;
    constexpr int WBLK = KT * MT * 512;          // halves per tap block
    constexpr int NBUF = (TAPS > 1) ? 2 : 1;
    extern __shared__ _Float16 smem[];
    _Float16* sIn = smem;
    _Float16* swt = smem + INH;
    const int b = blockIdx.x, tid = threadIdx.x;
    const uint32_t swtOff = (uint32_t)(uintptr_t)swt;  // LDS byte address

    // async stage of one tap's A-fragments into LDS buffer bufSel
    auto stageTap = [&](int tap, int bufSel) {
        const uint64_t g0 = (uint64_t)(uintptr_t)(wf + (size_t)tap * WBLK);
        const uint32_t l0 = swtOff + (uint32_t)bufSel * (WBLK * 2);
        constexpr int CH16 = (WBLK * 2) / 16;    // b128 chunks
        for (int i = tid; i < CH16; i += 256)
            async_g2l_b128(l0 + (uint32_t)i * 16, g0 + (uint64_t)i * 16);
    };

    {   // zero padded input region (covers spatial pad + channel pad)
        uint32_t* z = (uint32_t*)sIn;
        for (int i = tid; i < NPOS * CHS / 2; i += 256) z[i] = 0u;
    }
    stageTap(0, 0);          // overlap first weight block with input staging
    __syncthreads();         // zeroing complete
    const float* inb = in + (size_t)b * CIN * 121;
    for (int i = tid; i < CIN * 121; i += 256) {
        int c = i / 121, p = i - c * 121;
        float v = inb[c * 121 + p];
        if (st) v = v * st[c] + st[CIN + c];
        v = aact(v, act);
        int y = p / 11, xx = p - y * 11;
        sIn[((y + PAD) * WP + (xx + PAD)) * CHS + c] = (_Float16)v;
    }
    wait_async0();           // weight buffer 0 resident
    __syncthreads();         // input + weights visible to all waves

    const int lane = tid & 31, wv = tid >> 5;
    const int n = lane & 15, hi = lane >> 4;
    const int s = wv * 16 + n;
    const int se = s < 121 ? s : 120;
    const int y0 = se / 11, x0 = se - y0 * 11;

    v8f acc[MT];
#pragma unroll
    for (int m = 0; m < MT; ++m)
#pragma unroll
        for (int r = 0; r < 8; ++r) acc[m][r] = 0.f;

    for (int tap = 0; tap < TAPS; ++tap) {
        if (tap + 1 < TAPS) stageTap(tap + 1, (tap + 1) & 1);  // async prefetch
        const _Float16* wcur = swt + (size_t)(NBUF > 1 ? (tap & 1) : 0) * WBLK;
        const int dy = tap / KSZ, dx = tap - dy * KSZ;
        const _Float16* bp = sIn + ((y0 + dy) * WP + (x0 + dx)) * CHS + hi * 8;
#pragma unroll
        for (int kt = 0; kt < KT; ++kt) {
            V16 bu;  // B frag: K = 8*hi + (h%8) + 16*(h/8), N = lane%16
            bu.h[0] = *(const v8h*)(bp + kt * 32);
            bu.h[1] = *(const v8h*)(bp + kt * 32 + 16);
            const _Float16* ap = wcur + (size_t)kt * MT * 512 + lane * 16;
#pragma unroll
            for (int mt = 0; mt < MT; ++mt) {
                V16 au;
                au.h[0] = *(const v8h*)(ap + mt * 512);
                au.h[1] = *(const v8h*)(ap + mt * 512 + 8);
                acc[mt] = __builtin_amdgcn_wmma_f32_16x16x32_f16(
                    false, au.v, false, bu.v, (short)0, acc[mt], false, false);
            }
        }
        if (tap + 1 < TAPS) {
            wait_async0();   // next tap block resident (this wave's loads)
            __syncthreads(); // and everyone done reading the buffer we refill next
        }
    }

    if (s < 121) {
        float* ob = out + (size_t)b * COUT * 121;
#pragma unroll
        for (int mt = 0; mt < MT; ++mt)
#pragma unroll
            for (int r = 0; r < 8; ++r) {
                int m = mt * 16 + hi * 8 + r;
                if (m < COUT) ob[m * 121 + s] = acc[mt][r];
            }
    }
}

constexpr int conv_smem_bytes(int CIN, int COUT, int KSZ, int PAD, int CHS) {
    int WP = 11 + 2 * PAD, NPOS = WP * WP;
    int KT = (CIN + 31) / 32, MT = (COUT + 15) / 16;
    int a = ((NPOS * CHS) + 63) & ~63;
    int nbuf = (KSZ * KSZ > 1) ? 2 : 1;
    return (a + nbuf * KT * MT * 512) * 2;
}

// ---------------------------------------------------------------------------
// out5: dynamic per-sample 3x3 conv with gathered weights
// spec[b,j,k,r,t] = att[4-j,c] * x[4-j,c,4+r,4+t], c = idx[b,4-j,k]
// out5[b,j,p] = sum_k sum_{r,t} spec * H[b,k, y+r-1, x+t-1]
// ---------------------------------------------------------------------------
__global__ __launch_bounds__(128) void out5_kernel(
    const float* __restrict__ x, const float* __restrict__ att,
    const int* __restrict__ idx, const float* __restrict__ H,
    float* __restrict__ out)
{
    extern __shared__ float smf[];
    float* w  = smf;            // 5*100*9 = 4500
    float* hp = smf + 4500;     // 100 * 13*13 = 16900 (zero-padded)
    const int b = blockIdx.x, tid = threadIdx.x;
    const uint32_t hpOff = (uint32_t)(uintptr_t)hp;
    for (int i = tid; i < 100 * 169; i += 128) hp[i] = 0.f;
    __syncthreads();
    // async scatter-copy of H tile into padded LDS image
    for (int i = tid; i < 100 * 121; i += 128) {
        int c = i / 121, p = i - c * 121;
        uint32_t l = hpOff + (uint32_t)(c * 169 + (p / 11 + 1) * 13 + (p % 11 + 1)) * 4;
        uint64_t g = (uint64_t)(uintptr_t)(H + ((size_t)b * 100 + c) * 121 + p);
        async_g2l_b32(l, g);
    }
    for (int i = tid; i < 4500; i += 128) {
        int j = i / 900, rem = i - j * 900;
        int k = rem / 9, r = (rem % 9) / 3, t = rem % 3;
        int qb = 4 - j;
        int c = idx[(size_t)b * 500 + qb * 100 + k];
        w[i] = att[qb * 200 + c] * x[((size_t)qb * 200 + c) * 121 + (4 + r) * 11 + (4 + t)];
    }
    wait_async0();
    __syncthreads();
    if (tid < 121) {
        const int y = tid / 11, xx = tid - y * 11;
        for (int j = 0; j < 5; ++j) {
            float accv = 0.f;
            const float* wj = w + j * 900;
            for (int k = 0; k < 100; ++k) {
                const float* hb = hp + k * 169 + y * 13 + xx;
                const float* wk = wj + k * 9;
#pragma unroll
                for (int r = 0; r < 3; ++r)
#pragma unroll
                    for (int t = 0; t < 3; ++t)
                        accv += wk[r * 3 + t] * hb[r * 13 + t];
            }
            out[((size_t)b * 5 + j) * 121 + tid] = accv;
        }
    }
}

// ---------------------------------------------------------------------------
// Final head: g = mean(A, spatial); out = g @ out_w.T + out_b
// ---------------------------------------------------------------------------
__global__ __launch_bounds__(128) void final_kernel(
    const float* __restrict__ A, const float* __restrict__ W,
    const float* __restrict__ bias, float* __restrict__ out)
{
    __shared__ float g[100];
    const int b = blockIdx.x, tid = threadIdx.x;
    if (tid < 100) {
        const float* p = A + ((size_t)b * 100 + tid) * 121;
        float s = 0.f;
        for (int i = 0; i < 121; ++i) s += p[i];
        g[tid] = s * (1.f / 121.f);
    }
    __syncthreads();
    if (tid < 16) {
        float s = bias[tid];
        const float* wr = W + tid * 100;
        for (int c = 0; c < 100; ++c) s += g[c] * wr[c];
        out[(size_t)b * 16 + tid] = s;
    }
}

// ---------------------------------------------------------------------------
#define LAUNCH_CONV(CI, CO, KS, PD, CH, inp, stp, act, wfp, outp)                        \
    do {                                                                                 \
        constexpr int _sm = conv_smem_bytes(CI, CO, KS, PD, CH);                         \
        hipFuncSetAttribute(reinterpret_cast<const void*>(&conv_wmma<CI, CO, KS, PD, CH>), \
                            hipFuncAttributeMaxDynamicSharedMemorySize, _sm);            \
        conv_wmma<CI, CO, KS, PD, CH><<<dim3(512), dim3(256), _sm, stream>>>(            \
            inp, stp, act, wfp, outp);                                                   \
    } while (0)

extern "C" void kernel_launch(void* const* d_in, const int* in_sizes, int n_in,
                              void* d_out, int out_size, void* d_ws, size_t ws_size,
                              hipStream_t stream) {
    const float* x     = (const float*)d_in[0];
    const float* kw1   = (const float*)d_in[1];
    const float* kw2   = (const float*)d_in[2];
    const float* kw3   = (const float*)d_in[3];
    const float* kw4   = (const float*)d_in[4];
    const float* c3w   = (const float*)d_in[5];
    const float* c3b   = (const float*)d_in[6];
    const float* c5w   = (const float*)d_in[7];
    const float* c5b   = (const float*)d_in[8];
    const float* c7w   = (const float*)d_in[9];
    const float* c7b   = (const float*)d_in[10];
    const float* fc1w  = (const float*)d_in[11];
    const float* fc1b  = (const float*)d_in[12];
    const float* fc2w  = (const float*)d_in[13];
    const float* fc2b  = (const float*)d_in[14];
    const float* e5w   = (const float*)d_in[15];
    const float* e100w = (const float*)d_in[17];
    const float* conv1w= (const float*)d_in[19];
    const float* l1w   = (const float*)d_in[21];
    const float* l1ew  = (const float*)d_in[23];
    const float* lws   = (const float*)d_in[25];
    const float* outw  = (const float*)d_in[27];
    const float* outb  = (const float*)d_in[28];
    const int*   idx   = (const int*)d_in[29];
    float* dout = (float*)d_out;

    // ---- workspace layout (floats) ----
    float* ws = (float*)d_ws;
    size_t o = 0;
    float* ATT   = ws + o; o += 512 * 200;
    float* STATS = ws + o; o += 15 * 256;     // 15 slots: [scale C][bias C]
    float* BUF0  = ws + o; o += (size_t)512 * 100 * 121;
    float* BUF1  = ws + o; o += (size_t)512 * 100 * 121;
    float* BUF2  = ws + o; o += (size_t)512 * 100 * 121;
    float* CBUF  = ws + o; o += (size_t)512 * 50 * 121;
    float* SB0   = ws + o; o += (size_t)512 * 5 * 121;
    float* SB1   = ws + o; o += (size_t)512 * 5 * 121;
    float* SB2   = ws + o; o += (size_t)512 * 5 * 121;
    _Float16* WF = (_Float16*)(ws + o);
    size_t ho = 0;
    auto halloc = [&](size_t nh) { _Float16* p = WF + ho; ho += nh; return p; };
    auto fragHalves = [](int CO, int CI, int KS) {
        return (size_t)KS * KS * ((CI + 31) / 32) * ((CO + 15) / 16) * 512;
    };
    _Float16* WKW1 = halloc(fragHalves(100, 200, 5));
    _Float16* WKW2 = halloc(fragHalves(100, 100, 5));
    _Float16* WKW3 = halloc(fragHalves(100, 100, 5));
    _Float16* WKW4 = halloc(fragHalves(100, 100, 5));
    _Float16* WL0  = halloc(fragHalves(100, 100, 5));
    _Float16* WL1  = halloc(fragHalves(100, 100, 5));
    _Float16* WL2  = halloc(fragHalves(100, 100, 5));
    _Float16* WL3  = halloc(fragHalves(100, 100, 5));
    _Float16* WLL1 = halloc(fragHalves(100, 50, 5));
    _Float16* WC1  = halloc(fragHalves(50, 5, 3));
    _Float16* WE100= halloc(fragHalves(100, 200, 1));
    _Float16* WE5  = halloc(fragHalves(5, 200, 1));
    _Float16* WL1E = halloc(fragHalves(100, 50, 1));

    float* S1   = STATS + 0 * 256;
    float* S2   = STATS + 1 * 256;
    float* S3   = STATS + 2 * 256;
    float* S4   = STATS + 3 * 256;
    float* SSP  = STATS + 4 * 256;
    float* SB5  = STATS + 5 * 256;
    float* SORI = STATS + 6 * 256;
    float* SXA  = STATS + 7 * 256;
    float* S5C  = STATS + 8 * 256;
    float* SL1  = STATS + 9 * 256;
    float* SE1  = STATS + 10 * 256;
    float* SJ[4] = {STATS + 11 * 256, STATS + 12 * 256, STATS + 13 * 256, STATS + 14 * 256};

    auto swz = [&](const float* W, _Float16* dst, int CO, int CI, int KS) {
        int KT = (CI + 31) / 32, MT = (CO + 15) / 16;
        int total = KS * KS * KT * MT * 512;
        swz_kernel<<<dim3((total + 255) / 256), dim3(256), 0, stream>>>(
            W, dst, CO, CI, KS, KT, MT, total);
    };
    auto stats = [&](const float* buf, int C, float* slot) {
        stats_kernel<<<dim3(C), dim3(256), 0, stream>>>(buf, C, slot);
    };
    auto combine = [&](const float* a, const float* sa, const float* b2, const float* sb,
                       float* outp, int C, int aA, int aB, int aF) {
        int total = 512 * C * 121;
        combine_kernel<<<dim3((total + 255) / 256), dim3(256), 0, stream>>>(
            a, sa, b2, sb, outp, C, total, aA, aB, aF);
    };

    // ---- weight swizzles ----
    swz(kw1, WKW1, 100, 200, 5);
    swz(kw2, WKW2, 100, 100, 5);
    swz(kw3, WKW3, 100, 100, 5);
    swz(kw4, WKW4, 100, 100, 5);
    swz(lws + 0 * 100 * 100 * 25, WL0, 100, 100, 5);
    swz(lws + 1 * 100 * 100 * 25, WL1, 100, 100, 5);
    swz(lws + 2 * 100 * 100 * 25, WL2, 100, 100, 5);
    swz(lws + 3 * 100 * 100 * 25, WL3, 100, 100, 5);
    swz(l1w,    WLL1, 100, 50, 5);
    swz(conv1w, WC1, 50, 5, 3);
    swz(e100w,  WE100, 100, 200, 1);
    swz(e5w,    WE5, 5, 200, 1);
    swz(l1ew,   WL1E, 100, 50, 1);

    // ---- attention ----
    attention_kernel<<<dim3(512), dim3(256), 0, stream>>>(
        x, c3w, c3b, c5w, c5b, c7w, c7b, fc1w, fc1b, fc2w, fc2b, ATT);

    // ---- trunk: 4x (conv5x5 -> bn -> lrelu), bn folded into next staging ----
    LAUNCH_CONV(200, 100, 5, 2, 232, x, (const float*)nullptr, 0, WKW1, BUF0);
    stats(BUF0, 100, S1);
    LAUNCH_CONV(100, 100, 5, 2, 136, BUF0, S1, 2, WKW2, BUF1);
    stats(BUF1, 100, S2);
    LAUNCH_CONV(100, 100, 5, 2, 136, BUF1, S2, 2, WKW3, BUF2);
    stats(BUF2, 100, S3);
    LAUNCH_CONV(100, 100, 5, 2, 136, BUF2, S3, 2, WKW4, BUF0);
    stats(BUF0, 100, S4);
    // spact = lrelu(bn(c1x1(x, extra100)))
    LAUNCH_CONV(200, 100, 1, 0, 232, x, (const float*)nullptr, 0, WE100, BUF1);
    stats(BUF1, 100, SSP);
    // H = lrelu( lrelu(bn(C4)) + lrelu(bn(SP)) )
    combine(BUF0, S4, BUF1, SSP, BUF2, 100, 2, 2, 2);

    // ---- out5 (dynamic per-sample conv) + origin + double-bn merge ----
    {
        int smB = (4500 + 100 * 169) * 4;
        hipFuncSetAttribute(reinterpret_cast<const void*>(&out5_kernel),
                            hipFuncAttributeMaxDynamicSharedMemorySize, smB);
        out5_kernel<<<dim3(512), dim3(128), smB, stream>>>(x, ATT, idx, BUF2, SB0);
    }
    stats(SB0, 5, SB5);
    LAUNCH_CONV(200, 5, 1, 0, 232, x, (const float*)nullptr, 0, WE5, SB1);
    stats(SB1, 5, SORI);
    combine(SB0, SB5, SB1, SORI, SB2, 5, 0, 0, 0);  // XAraw = bn(out5)+bn(origin)
    stats(SB2, 5, SXA);

    // ---- conv1 3x3 on relu(bn(XAraw)) ----
    LAUNCH_CONV(5, 50, 3, 1, 40, SB2, SXA, 1, WC1, CBUF);
    stats(CBUF, 50, S5C);

    // ---- layer l1 (50->100 w/ 1x1 bn shortcut) ----
    LAUNCH_CONV(50, 100, 5, 2, 72, CBUF, S5C, 1, WLL1, BUF0);
    stats(BUF0, 100, SL1);
    LAUNCH_CONV(50, 100, 1, 0, 72, CBUF, S5C, 1, WL1E, BUF1);
    stats(BUF1, 100, SE1);
    combine(BUF1, SE1, BUF0, SL1, BUF2, 100, 0, 1, 1);  // A1

    // ---- 4 residual layers ----
    const _Float16* WJ[4] = {WL0, WL1, WL2, WL3};
    float* cur = BUF2;
    float* spare = BUF1;
    for (int j = 0; j < 4; ++j) {
        LAUNCH_CONV(100, 100, 5, 2, 136, cur, (const float*)nullptr, 0, WJ[j], BUF0);
        stats(BUF0, 100, SJ[j]);
        combine(cur, (const float*)nullptr, BUF0, SJ[j], spare, 100, 0, 1, 1);
        float* t = cur; cur = spare; spare = t;
    }

    // ---- head ----
    final_kernel<<<dim3(512), dim3(128), 0, stream>>>(cur, outw, outb, dout);
}
